// WeatherGNN_29712583754331
// MI455X (gfx1250) — compile-verified, matrix operand order checked by reading, not run on hardware
//
#include <hip/hip_runtime.h>
#include <hip/hip_bf16.h>
#include <math.h>

typedef __bf16 bf16;
typedef __attribute__((ext_vector_type(16))) __bf16 v16bf;
typedef __attribute__((ext_vector_type(8)))  __bf16 v8bf;
typedef __attribute__((ext_vector_type(8)))  float  v8f;

// Problem constants (match reference)
// B=4, T=7, N1=4096, FN=8, HID=16, D=128, N2=256, C=16, OUT=5, K2=8

__device__ __forceinline__ bf16 f2bf(float x){
  unsigned u = __builtin_bit_cast(unsigned, x);
  u += 0x7fffu + ((u >> 16) & 1u);          // round-to-nearest-even
  unsigned short h = (unsigned short)(u >> 16);
  return __builtin_bit_cast(bf16, h);
}

// 16-bit A/B fragment (16x32 tile slice for this lane), per ISA 7.12.2:
// lanes 0-15 row M=lane hold K = {0..7, 16..23}; lanes 16-31 same M hold K = {8..15, 24..31}
__device__ __forceinline__ v16bf load_frag(const bf16* rowPtr, int kk, int half){
  v8bf c0 = *(const v8bf*)(rowPtr + kk + half*8);
  v8bf c1 = *(const v8bf*)(rowPtr + kk + 16 + half*8);
  v16bf f;
  #pragma unroll
  for (int e = 0; e < 8; ++e){ f[e] = c0[e]; f[8+e] = c1[e]; }
  return f;
}

// ---------------------------------------------------------------- prep
__global__ void k_prep(const float* FE, const float* bias_pool, const float* weights_pool,
                       float* S, float* biasF, float* wts, float* A2d){
  int tid = threadIdx.x;
  __shared__ float L[64];
  if (tid < 64){
    int n = tid >> 3, m = tid & 7;
    float a = 0.f;
    for (int d = 0; d < 16; ++d) a += FE[n*16+d]*FE[m*16+d];
    L[tid] = a > 0.f ? a : 0.f;
  }
  __syncthreads();
  if (tid < 8){
    float mx = -1e30f;
    for (int m = 0; m < 8; ++m) mx = fmaxf(mx, L[tid*8+m]);
    float e[8]; float s = 0.f;
    for (int m = 0; m < 8; ++m){ e[m] = expf(L[tid*8+m]-mx); s += e[m]; }
    for (int m = 0; m < 8; ++m) S[tid*8+m] = e[m]/s;
  }
  if (tid < 128){
    int n = tid >> 4, o = tid & 15;
    float a = 0.f;
    for (int d = 0; d < 16; ++d) a += FE[n*16+d]*bias_pool[d*16+o];
    biasF[tid] = a;
  }
  // wts[n][k][i][o] = sum_d FE[n][d]*weights_pool[d][k][i][o]
  for (int idx = tid; idx < 4096; idx += 256){
    int n = idx >> 9, rem = idx & 511;
    float a = 0.f;
    for (int d = 0; d < 16; ++d) a += FE[n*16+d]*weights_pool[d*512 + rem];
    wts[idx] = a;
  }
  for (int idx = tid; idx < 65536; idx += 256) A2d[idx] = 0.f;
}

// ------------------------------------------- transpose weights -> bf16
__global__ void k_wt(const float* Wq, const float* Wk, const float* Wg, const float* Wu,
                     bf16* WqT, bf16* WkT, bf16* WgT, bf16* WuT){
  int id = blockIdx.x*256 + threadIdx.x;       // 65536 total
  int w = id >> 14, rem = id & 16383;
  int n = rem >> 7, k = rem & 127;
  const float* src = (w==0)?Wq:(w==1)?Wk:(w==2)?Wg:Wu;
  bf16* dst = (w==0)?WqT:(w==1)?WkT:(w==2)?WgT:WuT;
  dst[n*128+k] = f2bf(src[k*128+n]);
}

// ---------------------------------------------------------------- feat1
__global__ void __launch_bounds__(128) k_feat1(
    const float* x, const float* W_embed, const float* b_embed,
    const float* S, const float* wts, const float* biasF,
    float* feat1, bf16* feat1h){
  int bq = blockIdx.x;                         // 0..16383
  int b = bq >> 12, q = bq & 4095;
  int tid = threadIdx.x;                       // 128
  int n = tid >> 4, c = tid & 15;
  __shared__ float h0[128], mx[128];
  float a = b_embed[c];
  for (int t = 0; t < 7; ++t){
    float xv = x[(((size_t)(b*7 + t))*4096 + q)*8 + n];
    a += xv * W_embed[t*16 + c];
  }
  h0[tid] = a;
  __syncthreads();
  float m = 0.f;
  for (int mm = 0; mm < 8; ++mm) m += S[n*8+mm]*h0[mm*16 + c];
  mx[tid] = m;
  __syncthreads();
  float hf = biasF[tid];
  const float* w0 = wts + n*512;
  const float* w1 = wts + n*512 + 256;
  for (int i = 0; i < 16; ++i)
    hf += h0[n*16+i]*w0[i*16+c] + mx[n*16+i]*w1[i*16+c];
  size_t off = (size_t)bq*128 + tid;
  feat1[off] = hf;
  feat1h[off] = f2bf(hf);
}

// ---------------------------------------------------------------- feat2
__global__ void k_feat2(const float* feat1, float* feat2){
  int idx = blockIdx.x*256 + threadIdx.x;      // 4*256*128 = 131072
  int d = idx & 127;
  int m = (idx >> 7) & 255;
  int b = idx >> 15;
  const float* src = feat1 + ((size_t)(b*4096 + m*16))*128 + d;
  float a = 0.f;
  for (int j = 0; j < 16; ++j) a += src[(size_t)j*128];
  feat2[idx] = a;
}

// ------------------------------------------------- Q/K projection (WMMA)
__global__ void __launch_bounds__(256) k_qk(
    const bf16* feat1h, const bf16* WqT, const bf16* WkT,
    const float* bq, const float* bk, bf16* qh, bf16* kh){
  int rb = blockIdx.x;                          // 1024 row blocks of 16 over 16384 rows
  int which = blockIdx.y;
  const bf16* Bt = which ? WkT : WqT;
  const float* bias = which ? bk : bq;
  bf16* dst = which ? kh : qh;
  int tid = threadIdx.x;
  int wave = tid >> 5, lane = tid & 31;
  int r = lane & 15, half = lane >> 4;
  int cb = wave;                                // 8 waves -> 8 col blocks
  const bf16* arow = feat1h + ((size_t)(rb*16 + r))*128;
  const bf16* brow = Bt + ((size_t)(cb*16 + r))*128;
  v8f acc = {};
  #pragma unroll
  for (int kk = 0; kk < 128; kk += 32){
    v16bf af = load_frag(arow, kk, half);
    v16bf bf_ = load_frag(brow, kk, half);
    acc = __builtin_amdgcn_wmma_f32_16x16x32_bf16(false, af, false, bf_,
                                                  (short)0, acc, false, false);
  }
  int ncol = cb*16 + r;
  int mbase = half*8;
  #pragma unroll
  for (int v = 0; v < 8; ++v){
    int rowg = rb*16 + mbase + v;
    dst[(size_t)rowg*128 + ncol] = f2bf(acc[v] + bias[ncol]);
  }
}

// -------------------- attention: full row tile in LDS, block sums -> A2d
__global__ void __launch_bounds__(256) k_attn(const bf16* qh, const bf16* kh, float* A2d){
  extern __shared__ float smem[];
  float* sc = smem;                   // 16*4096 scores
  float* red = smem + 16*4096;        // 256 reduction scratch
  float* rowstat = red + 256;         // 16: row max, then 1/sum
  int bi = blockIdx.x;
  int b = bi >> 8, I = bi & 255;
  int tid = threadIdx.x;
  int wave = tid >> 5, lane = tid & 31;
  int r = lane & 15, half = lane >> 4;
  const bf16* qrow = qh + ((size_t)(b*4096 + I*16 + r))*128;
  const float scale = 0.088388347648318447f;    // 1/sqrt(128)
  for (int jb = wave; jb < 256; jb += 8){
    const bf16* krow = kh + ((size_t)(b*4096 + jb*16 + r))*128;
    v8f acc = {};
    #pragma unroll
    for (int kk = 0; kk < 128; kk += 32){
      v16bf af = load_frag(qrow, kk, half);
      v16bf bf_ = load_frag(krow, kk, half);
      acc = __builtin_amdgcn_wmma_f32_16x16x32_bf16(false, af, false, bf_,
                                                    (short)0, acc, false, false);
    }
    int mbase = half*8;
    #pragma unroll
    for (int v = 0; v < 8; ++v)
      sc[(size_t)(mbase+v)*4096 + jb*16 + r] = acc[v]*scale;
  }
  __syncthreads();
  int rr = tid >> 4, sub = tid & 15;
  float mxv = -1e30f;
  for (int j = sub; j < 4096; j += 16) mxv = fmaxf(mxv, sc[rr*4096 + j]);
  red[tid] = mxv;
  __syncthreads();
  if (sub == 0){
    float m = -1e30f;
    for (int s = 0; s < 16; ++s) m = fmaxf(m, red[rr*16+s]);
    rowstat[rr] = m;
  }
  __syncthreads();
  float rm = rowstat[rr];
  float sm = 0.f;
  for (int j = sub; j < 4096; j += 16){
    float e = expf(sc[rr*4096 + j] - rm);
    sc[rr*4096 + j] = e;
    sm += e;
  }
  red[tid] = sm;
  __syncthreads();
  if (sub == 0){
    float s = 0.f;
    for (int ss = 0; ss < 16; ++ss) s += red[rr*16+ss];
    rowstat[rr] = 1.f / s;
  }
  __syncthreads();
  // tid == column block q; A2d[I][q] += (1/B) * sum_{r,j in block} attn
  float acc = 0.f;
  for (int r2 = 0; r2 < 16; ++r2){
    float inv = rowstat[r2];
    const float* row = sc + (size_t)r2*4096 + tid*16;
    float ps = 0.f;
    #pragma unroll
    for (int jj = 0; jj < 16; ++jj) ps += row[jj];
    acc += ps*inv;
  }
  atomicAdd(&A2d[I*256 + tid], acc*0.25f);
}

// --------------------------- fused m1/m2/agg/upd/out per (b, block i)
__global__ void __launch_bounds__(128) k_final(
    const float* feat1, const float* feat2, const float* A2d,
    const float* a1, const int* nbr2,
    const bf16* WgT, const float* bg,
    const float* Wa, const float* ba,
    const bf16* WuT, const float* bu,
    const float* Wd, const float* bd,
    float* out){
  int bi = blockIdx.x;                          // 0..1023
  int b = bi >> 8, i = bi & 255;
  int tid = threadIdx.x;                        // 128 = 4 waves
  int wave = tid >> 5, lane = tid & 31;
  int r = lane & 15, half = lane >> 4;

  __shared__ float sf[16*128];
  __shared__ float Gs[256];
  __shared__ __align__(16) bf16 tbh[16*128];
  __shared__ float m1s[16*128];
  __shared__ float m2s[128];
  __shared__ float aggv[16];
  __shared__ float aggc;
  __shared__ __align__(16) bf16 uh[16*128];
  __shared__ float updf[16*128];

  for (int it = 0; it < 16; ++it)
    sf[it*128 + tid] = feat1[((size_t)(b*4096 + i*16 + it))*128 + tid];
  Gs[tid]       = a1[((size_t)(i*16 + (tid>>4)))*4096 + i*16 + (tid&15)];
  Gs[128 + tid] = a1[((size_t)(i*16 + ((128+tid)>>4)))*4096 + i*16 + (tid&15)];
  __syncthreads();

  { // t[k][d] = sum_n G[n][k]*sf[n][d]  (K=16 too small for WMMA; scalar)
    int d = tid;
    for (int k = 0; k < 16; ++k){
      float a = 0.f;
      for (int n = 0; n < 16; ++n) a += Gs[n*16+k]*sf[n*128+d];
      tbh[k*128+d] = f2bf(a);
    }
  }
  { // m2[d] = sum_k2 A2d[i, nbr2[i,k2]] * feat2[b, nbr2[i,k2], d]
    int d = tid;
    float a = 0.f;
    for (int k2 = 0; k2 < 8; ++k2){
      int nb = nbr2[i*8+k2];
      a += A2d[i*256 + nb] * feat2[((size_t)(b*256 + nb))*128 + d];
    }
    m2s[d] = a;
  }
  __syncthreads();

  // m1 = relu(t @ Wg + bg)  via WMMA: 4 waves x 2 col blocks
  for (int cbi = 0; cbi < 2; ++cbi){
    int cb = wave + cbi*4;
    const bf16* arow = tbh + r*128;
    const bf16* brow = WgT + (size_t)(cb*16 + r)*128;
    v8f acc = {};
    #pragma unroll
    for (int kk = 0; kk < 128; kk += 32){
      v16bf af = load_frag(arow, kk, half);
      v16bf bf_ = load_frag(brow, kk, half);
      acc = __builtin_amdgcn_wmma_f32_16x16x32_bf16(false, af, false, bf_,
                                                    (short)0, acc, false, false);
    }
    int ncol = cb*16 + r, mbase = half*8;
    #pragma unroll
    for (int v = 0; v < 8; ++v){
      float xv = acc[v] + bg[ncol];
      m1s[(mbase+v)*128 + ncol] = xv > 0.f ? xv : 0.f;
    }
  }
  __syncthreads();

  if (tid == 0){
    float a = ba[0];
    for (int d = 0; d < 128; ++d) a += m2s[d]*(Wa[128+d] + Wa[256+d]);
    aggc = a;
  }
  __syncthreads();
  if (tid < 16){
    float a = aggc;
    for (int d = 0; d < 128; ++d) a += m1s[tid*128+d]*Wa[d];
    aggv[tid] = a > 0.f ? a : 0.f;
  }
  __syncthreads();
  for (int it = 0; it < 16; ++it)
    uh[it*128 + tid] = f2bf(sf[it*128 + tid] + aggv[it]);
  __syncthreads();

  // upd = relu((sf+agg) @ Wu + bu) via WMMA
  for (int cbi = 0; cbi < 2; ++cbi){
    int cb = wave + cbi*4;
    const bf16* arow = uh + r*128;
    const bf16* brow = WuT + (size_t)(cb*16 + r)*128;
    v8f acc = {};
    #pragma unroll
    for (int kk = 0; kk < 128; kk += 32){
      v16bf af = load_frag(arow, kk, half);
      v16bf bf_ = load_frag(brow, kk, half);
      acc = __builtin_amdgcn_wmma_f32_16x16x32_bf16(false, af, false, bf_,
                                                    (short)0, acc, false, false);
    }
    int ncol = cb*16 + r, mbase = half*8;
    #pragma unroll
    for (int v = 0; v < 8; ++v){
      float xv = acc[v] + bu[ncol];
      updf[(mbase+v)*128 + ncol] = xv > 0.f ? xv : 0.f;
    }
  }
  __syncthreads();

  if (tid < 80){
    int rr = tid/5, o = tid%5;
    float a = bd[o];
    for (int d = 0; d < 128; ++d) a += updf[rr*128+d]*Wd[d*5+o];
    out[((size_t)(b*4096 + i*16 + rr))*5 + o] = a;
  }
}

extern "C" void kernel_launch(void* const* d_in, const int* in_sizes, int n_in,
                              void* d_out, int out_size, void* d_ws, size_t ws_size,
                              hipStream_t stream){
  const float* x        = (const float*)d_in[0];
  const float* W_embed  = (const float*)d_in[1];
  const float* b_embed  = (const float*)d_in[2];
  const float* FE       = (const float*)d_in[3];
  const float* wpool    = (const float*)d_in[4];
  const float* bpool    = (const float*)d_in[5];
  const float* Wq = (const float*)d_in[6];
  const float* bq = (const float*)d_in[7];
  const float* Wk = (const float*)d_in[8];
  const float* bk = (const float*)d_in[9];
  const float* Wg = (const float*)d_in[10];
  const float* bg = (const float*)d_in[11];
  const float* Wa = (const float*)d_in[12];
  const float* ba = (const float*)d_in[13];
  const float* Wu = (const float*)d_in[14];
  const float* bu = (const float*)d_in[15];
  const float* Wd = (const float*)d_in[16];
  const float* bd = (const float*)d_in[17];
  // d_in[18]=s1, d_in[19]=s2: fixed group-of-16 selectors, folded analytically.
  const float* a1   = (const float*)d_in[20];
  const int*   nbr2 = (const int*)d_in[21];
  // d_in[22]=nbr3: m3 is dead in the reference (concat uses [m1,m2r,m2r]).
  float* out = (float*)d_out;
  (void)in_sizes; (void)n_in; (void)out_size; (void)ws_size;

  char* p = (char*)d_ws;
  auto alloc = [&](size_t bytes)->void*{
    void* r = p; p += (bytes + 255) & ~(size_t)255; return r;
  };
  float* S     = (float*)alloc(64*4);
  float* biasF = (float*)alloc(128*4);
  float* wts   = (float*)alloc(4096*4);
  float* A2d   = (float*)alloc(65536*4);
  float* feat1 = (float*)alloc((size_t)4*4096*128*4);
  bf16*  feat1h= (bf16*) alloc((size_t)4*4096*128*2);
  float* feat2 = (float*)alloc((size_t)4*256*128*4);
  bf16*  qh    = (bf16*) alloc((size_t)4*4096*128*2);
  bf16*  kh    = (bf16*) alloc((size_t)4*4096*128*2);
  bf16*  WqT   = (bf16*) alloc(16384*2);
  bf16*  WkT   = (bf16*) alloc(16384*2);
  bf16*  WgT   = (bf16*) alloc(16384*2);
  bf16*  WuT   = (bf16*) alloc(16384*2);

  k_prep <<<1,    256, 0, stream>>>(FE, bpool, wpool, S, biasF, wts, A2d);
  k_wt   <<<256,  256, 0, stream>>>(Wq, Wk, Wg, Wu, WqT, WkT, WgT, WuT);
  k_feat1<<<16384,128, 0, stream>>>(x, W_embed, b_embed, S, wts, biasF, feat1, feat1h);
  k_feat2<<<512,  256, 0, stream>>>(feat1, feat2);
  k_qk   <<<dim3(1024,2), 256, 0, stream>>>(feat1h, WqT, WkT, bq, bk, qh, kh);

  size_t attn_lds = (size_t)(16*4096 + 256 + 16) * sizeof(float);   // ~263 KB of the 320 KB WGP LDS
  hipFuncSetAttribute(reinterpret_cast<const void*>(k_attn),
                      hipFuncAttributeMaxDynamicSharedMemorySize, (int)attn_lds);
  k_attn <<<1024, 256, attn_lds, stream>>>(qh, kh, A2d);

  k_final<<<1024, 128, 0, stream>>>(feat1, feat2, A2d, a1, nbr2,
                                    WgT, bg, Wa, ba, WuT, bu, Wd, bd, out);
}
